// reservoir_computing_87179246174804
// MI455X (gfx1250) — matrix operand order repaired and checked by reading, not compiled, not run on hardware
//
#include <hip/hip_runtime.h>
#include <hip/hip_bf16.h>

// ---------------- problem constants ----------------
#define NV 200      // N_V
#define DR 300      // D_R
#define LT 2048     // L_T
#define MT 13       // tiles along NV (208)
#define NT 19       // tiles along DR (304)
#define KT1 10      // K tiles for GEMM1 (K = DR padded to 320)
#define KT2 7       // K tiles for GEMM2 (K = NV padded to 224)
#define SK 320      // padded K (cols of S in LDS)
#define CK 224      // padded K (cols of C1^T in LDS)

typedef __attribute__((ext_vector_type(16))) __bf16 v16bf;
typedef __attribute__((ext_vector_type(8)))  __bf16 v8bf;
typedef __attribute__((ext_vector_type(8)))  float  v8f;

// LDS partition (bytes)
#define SS_OFF   0
#define SS_BYTES (208 * SK * 2)               // 133120
#define SC_OFF   (SS_OFF + SS_BYTES)          // 133120
#define SC_BYTES (304 * CK * 2)               // 136192
#define WIN_OFF  (SC_OFF + SC_BYTES)          // 269312
#define WIN_BYTES (304 * 4)                   // 1216
#define SX_OFF   (WIN_OFF + WIN_BYTES)        // 270528
#define SX_BYTES (2 * 208 * 4)                // 1664
#define SMEM_BYTES (SX_OFF + SX_BYTES)        // 272192  (<= 320KB WGP LDS)

// ---------------------------------------------------------------------------
// Prep: swizzle R (300x300 row-major f32) into bf16 B fragments (k order used here, same as A).
// B 32x16 (KxN), bf16: lane l -> col n = l&15, K half = l>>4, 16 contiguous K
// values per lane. Per tile: 32 lanes * 32 bytes = 1024B, lane-contiguous.
// ---------------------------------------------------------------------------
__global__ void resv_prep_R(const float* __restrict__ R, __bf16* __restrict__ Rf) {
    int tile = blockIdx.x;            // kt*NT + nt, grid = KT1*NT
    int lane = threadIdx.x;           // 32 threads
    int kt = tile / NT, nt = tile % NT;
    int n  = nt * 16 + (lane & 15);
    int kb = kt * 32 + (lane >> 4) * 16;
    __bf16* dst = Rf + (size_t)tile * 512 + lane * 16;
#pragma unroll
    for (int j = 0; j < 16; ++j) {
        int k = kb + j;
        float v = (k < DR && n < DR) ? R[k * DR + n] : 0.0f;
        dst[j] = (__bf16)v;
    }
}

// ---------------------------------------------------------------------------
// Prep: swizzle Input_network^T (INT[m][k] = IN[k][m]) into bf16 WMMA
// A fragments (k order used here, same as B). A 16x32 (MxK), bf16: lane l -> row m = l&15,
// 16 contiguous K = c..c+15 where c = (l<16)?0:16.
// ---------------------------------------------------------------------------
__global__ void resv_prep_IN(const float* __restrict__ IN, __bf16* __restrict__ INTf) {
    int tile = blockIdx.x;            // mt*KT2 + kt, grid = MT*KT2
    int lane = threadIdx.x;           // 32 threads
    int mt = tile / KT2, kt = tile % KT2;
    int m  = mt * 16 + (lane & 15);
    int c  = (lane < 16) ? 0 : 16;
    __bf16* dst = INTf + (size_t)tile * 512 + lane * 16;
#pragma unroll
    for (int j = 0; j < 8; ++j) {
        int k = kt * 32 + c + j;
        float v = (m < NV && k < NV) ? IN[k * NV + m] : 0.0f;
        dst[j] = (__bf16)v;
    }
#pragma unroll
    for (int j = 0; j < 8; ++j) {
        int k = kt * 32 + c + 8 + j;
        float v = (m < NV && k < NV) ? IN[k * NV + m] : 0.0f;
        dst[8 + j] = (__bf16)v;
    }
}

__device__ __forceinline__ float fast_tanh(float z) {
    z = fminf(fmaxf(z, -15.0f), 15.0f);
    float e = __expf(2.0f * z);
    return (e - 1.0f) / (e + 1.0f);
}

// ---------------------------------------------------------------------------
// Persistent single-WGP recurrence kernel: 1 block, 1024 threads = 32 waves.
// ---------------------------------------------------------------------------
__global__ void __launch_bounds__(1024, 1)
resv_recurrence(const float* __restrict__ train,     // (LT, NV)
                const float* __restrict__ Win,       // (1, DR)
                const float* __restrict__ p_delta,
                const float* __restrict__ p_rho,
                const float* __restrict__ p_b,
                const __bf16* __restrict__ Rf,       // GEMM1 B fragments
                const __bf16* __restrict__ INTf,     // GEMM2 A fragments
                float* __restrict__ Rstate)          // (LT, NV, DR)
{
    extern __shared__ char smem[];
    __bf16* sS   = (__bf16*)(smem + SS_OFF);   // state, bf16 [208][SK]
    __bf16* sC1  = (__bf16*)(smem + SC_OFF);   // rho*(S@R) transposed, bf16 [304][CK]
    float*  sWin = (float*)(smem + WIN_OFF);   // [304]
    float*  sX   = (float*)(smem + SX_OFF);    // [2][208] x_t staging

    const int tidx = threadIdx.x;
    const int lane = tidx & 31;
    const int wave = tidx >> 5;
    const float delta = p_delta[0];
    const float rho   = p_rho[0];
    const float bb    = p_b[0];

    // zero-init state and C1 (kills NaN risk in padded regions), load constants
    for (int i = tidx; i < 208 * SK; i += 1024) sS[i] = (__bf16)0.0f;
    for (int i = tidx; i < 304 * CK; i += 1024) sC1[i] = (__bf16)0.0f;
    for (int i = tidx; i < 304; i += 1024) sWin[i] = (i < DR) ? Win[i] : 0.0f;
    for (int i = tidx; i < 2 * 208; i += 1024) {
        int buf = i / 208, j = i % 208;
        sX[i] = (buf == 0 && j < NV) ? train[j] : 0.0f;
    }
    __syncthreads();

    const int cofs = (lane < 16) ? 0 : 16;       // A-frag K sub-offset
    const int nloc = lane & 15;
    const int half = lane >> 4;

    for (int t = 0; t < LT; ++t) {
        const float* xcur = sX + (t & 1) * 208;

        // Overlap: async-prefetch x_{t+1} into the other staging buffer (ASYNCcnt)
        if (wave == 0 && t + 1 < LT) {
            for (int j = lane; j < NV; j += 32) {
                unsigned ldsa = (unsigned)(uintptr_t)(&sX[((t + 1) & 1) * 208 + j]);
                unsigned long long ga =
                    (unsigned long long)(uintptr_t)(train + (size_t)(t + 1) * NV + j);
                asm volatile("global_load_async_to_lds_b32 %0, %1, off"
                             :: "v"(ldsa), "v"(ga) : "memory");
            }
        }

        // ---- GEMM1: C1 = S(208x320) @ R(320x304), store rho*C1 as C1^T bf16 ----
        for (int tile = wave; tile < MT * NT; tile += 32) {
            int mt = tile / NT, nt = tile % NT;
            int m = mt * 16 + nloc;
            v8f acc = (v8f)0.0f;
            const __bf16* brow = Rf + (size_t)nt * 512 + lane * 16;
#pragma unroll
            for (int kt = 0; kt < KT1; ++kt) {
                const __bf16* arow = sS + m * SK + kt * 32 + cofs;
                v8bf a0 = *(const v8bf*)(arow);
                v8bf a1 = *(const v8bf*)(arow + 8);
                v16bf A = __builtin_shufflevector(a0, a1, 0,1,2,3,4,5,6,7,
                                                          8,9,10,11,12,13,14,15);
                v16bf B = *(const v16bf*)(brow + (size_t)kt * NT * 512);
                acc = __builtin_amdgcn_wmma_f32_16x16x32_bf16(
                          false, A, false, B, (short)0, acc, false, false);
            }
            // D layout: lane -> n = nt*16+nloc, rows m = mt*16 + half*8 + v
            int n  = nt * 16 + nloc;
            int mb = mt * 16 + half * 8;
            v8bf c;
#pragma unroll
            for (int v = 0; v < 8; ++v) c[v] = (__bf16)(rho * acc[v]);
            *(v8bf*)(sC1 + n * CK + mb) = c;   // one ds_store_b128
        }
        __syncthreads();

        // ---- GEMM2 + update: S_t = tanh(delta*x⊗win + INT@C1 + b) ----
        for (int tile = wave; tile < MT * NT; tile += 32) {
            int mt = tile / NT, nt = tile % NT;
            v8f acc = (v8f)0.0f;
            const __bf16* afrag = INTf + (size_t)mt * KT2 * 512 + lane * 16;
            const __bf16* bbase = sC1 + (nt * 16 + nloc) * CK + half * 16;
#pragma unroll
            for (int kt = 0; kt < KT2; ++kt) {
                v16bf A = *(const v16bf*)(afrag + (size_t)kt * 512);
                v16bf B = *(const v16bf*)(bbase + kt * 32);
                acc = __builtin_amdgcn_wmma_f32_16x16x32_bf16(
                          false, A, false, B, (short)0, acc, false, false);
            }
            int n   = nt * 16 + nloc;
            int mb  = mt * 16 + half * 8;
            float w = sWin[n];
#pragma unroll
            for (int v = 0; v < 8; ++v) {
                int m = mb + v;
                float z   = delta * xcur[m] * w + acc[v] + bb;
                float val = fast_tanh(z);
                sS[m * SK + n] = (__bf16)val;      // next-step state (bf16)
                if (m < NV && n < DR)               // streaming output, bypass cache
                    __builtin_nontemporal_store(
                        val, Rstate + ((size_t)t * NV + m) * DR + n);
            }
        }
        if (wave == 0 && t + 1 < LT)
            asm volatile("s_wait_asynccnt 0" ::: "memory");
        __syncthreads();
    }
}

// ---------------------------------------------------------------------------
// Readout: Pre[t][m] = dot(R_state[t][m][:], W_out[m][:]).  One wave per row,
// whole-GPU, bandwidth-bound (~491MB read @ 23.3 TB/s ≈ 21us).
// ---------------------------------------------------------------------------
__global__ void resv_readout(const float* __restrict__ Rstate,
                             const float* __restrict__ Wout,
                             float* __restrict__ Pre) {
    int row  = blockIdx.x * 8 + (threadIdx.x >> 5);  // row = t*NV + m
    int lane = threadIdx.x & 31;
    if (row >= LT * NV) return;
    int m = row % NV;
    const float* rs = Rstate + (size_t)row * DR;
    const float* wo = Wout + (size_t)m * DR;
    float s = 0.0f;
    for (int n = lane; n < DR; n += 32) s += rs[n] * wo[n];
#pragma unroll
    for (int off = 16; off; off >>= 1) s += __shfl_down(s, off);
    if (lane == 0) Pre[row] = s;
}

extern "C" void kernel_launch(void* const* d_in, const int* in_sizes, int n_in,
                              void* d_out, int out_size, void* d_ws, size_t ws_size,
                              hipStream_t stream) {
    const float* train = (const float*)d_in[0];   // (LT, NV)
    const float* Rnet  = (const float*)d_in[1];   // (DR, DR)
    const float* INnet = (const float*)d_in[2];   // (NV, NV)
    const float* Win   = (const float*)d_in[3];   // (1, DR)
    const float* Wout  = (const float*)d_in[4];   // (NV, DR)
    const float* dlt   = (const float*)d_in[5];
    const float* rho   = (const float*)d_in[6];
    const float* b     = (const float*)d_in[7];

    float* Rstate = (float*)d_out;                               // (LT,NV,DR)
    float* Pre    = (float*)d_out + (size_t)LT * NV * DR;        // (LT,NV)

    __bf16* Rf   = (__bf16*)d_ws;                                // KT1*NT tiles
    __bf16* INTf = (__bf16*)((char*)d_ws + (size_t)KT1 * NT * 1024);

    hipFuncSetAttribute((const void*)resv_recurrence,
                        hipFuncAttributeMaxDynamicSharedMemorySize, SMEM_BYTES);

    resv_prep_R <<<KT1 * NT, 32, 0, stream>>>(Rnet, Rf);
    resv_prep_IN<<<MT * KT2, 32, 0, stream>>>(INnet, INTf);
    resv_recurrence<<<1, 1024, SMEM_BYTES, stream>>>(
        train, Win, dlt, rho, b, Rf, INTf, Rstate);
    resv_readout<<<(LT * NV) / 8, 256, 0, stream>>>(Rstate, Wout, Pre);
}
